// MPNN_87076166959678
// MI455X (gfx1250) — matrix operand order repaired and checked by reading, not compiled, not run on hardware
//
#include <hip/hip_runtime.h>
#include <hip/hip_bf16.h>

// ---------------------------------------------------------------------------
// MPNN / GCN-style 3-layer message passing for MI455X (gfx1250, wave32).
// Edge MLP (800k x 128x128 GEMM per layer) dominates -> bf16 WMMA, fp32 acc.
// Weights are pre-packed into WMMA B-fragment order so each B-fragment is one
// 32-byte contiguous per-lane load; A-fragments come from LDS (ds_load_b128).
// ---------------------------------------------------------------------------

#define NN    50000      // nodes
#define EE    800000     // edges
#define DH    128        // hidden/feature dim
#define DEA   16         // edge-attr dim
#define LAY   3
#define LN_EPS 1e-5f

typedef __bf16 bf16_t;
typedef __attribute__((ext_vector_type(16))) __bf16 v16bf;
typedef __attribute__((ext_vector_type(8)))  __bf16 v8bf;
typedef __attribute__((ext_vector_type(4)))  __bf16 v4bf;
typedef __attribute__((ext_vector_type(8)))  float  v8f;

// ---------------------------------------------------------------------------
// WMMA layouts per CDNA5 ISA 7.12.2 (wave32).
// A (16x32 bf16, MxK): m = lane&15; lanes 0-15 hold K{0-7,16-23},
//   lanes 16-31 K{8-15,24-31}: elem j<8 -> k = kh*8 + (j<8? (j&7) : ...)
//   precisely: v=j>>1,h=j&1 -> k = (v/4)*16 + kh*8 + (v%4)*2 + h
// B (32x16 bf16, KxN): n = lane&15; same K order as A (table order),
//   elem j -> k = kb + j + (j>=8?8:0), kb = kh*8.  (pre-packed to contiguous 16 bf16 per lane)
// C/D (16x16 f32): VGPR r -> M = r + (lane>=16 ? 8:0), N = lane&15.
// ---------------------------------------------------------------------------
__device__ __forceinline__ v16bf load_a_frag(const bf16_t* __restrict__ src,
                                             int ld, int lane) {
  v16bf a;
  const int m  = lane & 15;
  const int kh = (lane >> 4) & 1;
#pragma unroll
  for (int j = 0; j < 16; ++j) {
    const int v = j >> 1, h = j & 1;
    const int k = ((v >> 2) << 4) + (kh << 3) + ((v & 3) << 1) + h;
    a[j] = src[m * ld + k];
  }
  return a;
}

#define WMMA_BF16(A, B, C) \
  __builtin_amdgcn_wmma_f32_16x16x32_bf16(false, (A), false, (B), (short)0, (C), false, false)

// ---------------------------------------------------------------------------
// Weight packing: P[frag][lane][16] where frag = (l*KC + kc)*8 + t and the 16
// bf16 per lane are k = kc*32 + (lane>=16?8:0) + j + (j>=8?8:0), n = t*16 + (lane&15).
// ---------------------------------------------------------------------------
__global__ void pack_b128_k(const float* __restrict__ W, bf16_t* __restrict__ P) {
  // W: [LAY][128][128] -> P: [LAY][4][8][32][16]
  const int idx = blockIdx.x * 256 + threadIdx.x;
  if (idx >= LAY * 4 * 8 * 32) return;
  const int lane = idx & 31;
  const int t    = (idx >> 5) & 7;
  const int kc   = (idx >> 8) & 3;
  const int l    = idx >> 10;
  const int n    = t * 16 + (lane & 15);
  const int kb   = kc * 32 + ((lane >> 4) & 1) * 8;
  const float* src = W + (size_t)l * DH * DH;
  bf16_t* dst = P + (size_t)idx * 16;
#pragma unroll
  for (int j = 0; j < 16; ++j)
    dst[j] = (bf16_t)src[(kb + j + ((j >> 3) << 3)) * DH + n];
}

__global__ void pack_b16_k(const float* __restrict__ W, bf16_t* __restrict__ P) {
  // W: [LAY][16][128] -> P: [LAY][8][32][16], zero padded for k >= 16
  const int idx = blockIdx.x * 256 + threadIdx.x;
  if (idx >= LAY * 8 * 32) return;
  const int lane = idx & 31;
  const int t    = (idx >> 5) & 7;
  const int l    = idx >> 8;
  const int n    = t * 16 + (lane & 15);
  const int kb   = ((lane >> 4) & 1) * 8;
  const float* src = W + (size_t)l * DEA * DH;
  bf16_t* dst = P + (size_t)idx * 16;
#pragma unroll
  for (int j = 0; j < 16; ++j) {
    const int k = kb + j + ((j >> 3) << 3);
    dst[j] = (k < DEA) ? (bf16_t)src[k * DH + n] : (bf16_t)0.0f;
  }
}

// ---------------------------------------------------------------------------
// GCN normalization: deg (f32 atomics) then dis = deg>0 ? rsqrt(deg) : 0
// ---------------------------------------------------------------------------
__global__ void degree_k(const int* __restrict__ col, float* __restrict__ deg) {
  const int e = blockIdx.x * 256 + threadIdx.x;
  if (e < EE) atomicAdd(&deg[col[e]], 1.0f);
}

__global__ void invsqrt_k(float* __restrict__ deg) {
  const int i = blockIdx.x * 256 + threadIdx.x;
  if (i < NN) {
    const float d = deg[i];
    deg[i] = (d > 0.0f) ? rsqrtf(d) : 0.0f;
  }
}

// ---------------------------------------------------------------------------
// Node GEMM: hx = h @ W  (N x 128 @ 128 x 128), bf16 WMMA, fp32 acc.
// Block = 256 thr = 8 waves; block owns 16 rows, wave w owns N cols [16w,16w+16)
// ---------------------------------------------------------------------------
__global__ void gemm_node_k(const float* __restrict__ h,
                            const v16bf* __restrict__ Wp,
                            float* __restrict__ hx) {
  __shared__ __align__(16) bf16_t As[16 * DH];   // 16 x 128 bf16 tile (4KB)
  const int mbase = blockIdx.x * 16;
  {
    const int i = threadIdx.x;                    // 8 consecutive elems each
    const float4 f0 = *(const float4*)(h + (size_t)mbase * DH + i * 8);
    const float4 f1 = *(const float4*)(h + (size_t)mbase * DH + i * 8 + 4);
    v8bf p;
    p[0] = (bf16_t)f0.x; p[1] = (bf16_t)f0.y; p[2] = (bf16_t)f0.z; p[3] = (bf16_t)f0.w;
    p[4] = (bf16_t)f1.x; p[5] = (bf16_t)f1.y; p[6] = (bf16_t)f1.z; p[7] = (bf16_t)f1.w;
    *(v8bf*)(As + i * 8) = p;
  }
  __syncthreads();

  const int wave = threadIdx.x >> 5, lane = threadIdx.x & 31;
  v8f acc = {};
#pragma unroll
  for (int kc = 0; kc < 4; ++kc) {
    const v16bf a = load_a_frag(As + kc * 32, DH, lane);
    const v16bf b = Wp[(kc * 8 + wave) * 32 + lane];     // 32B contiguous/lane
    acc = WMMA_BF16(a, b, acc);
  }
  const int n = lane & 15, mh = (lane >> 4) & 1;
#pragma unroll
  for (int r = 0; r < 8; ++r)
    hx[(size_t)(mbase + r + mh * 8) * DH + wave * 16 + n] = acc[r];
}

// ---------------------------------------------------------------------------
// Fused edge pipeline. Block = 8 waves; wave owns 16 edges.
//  stage0: gather hx[row] rows into LDS (coalesced float4, stored bf16)
//  stage1: t = relu(edge_attr @ W1 + b1)    (K=16 zero-padded to 32, 8 WMMA)
//  stage2: ea = t @ W2                      (LDS repack D->A layout, 32 WMMA)
//  stage3: msg = hx_row*w + ea + b2 ; atomicAdd into agg[col]
// ---------------------------------------------------------------------------
__global__ void __launch_bounds__(256)
edge_fused_k(const float* __restrict__ ea,
             const v16bf* __restrict__ W1p,
             const float* __restrict__ b1,
             const v16bf* __restrict__ W2p,
             const float* __restrict__ b2,
             const int* __restrict__ eidx,
             const float* __restrict__ dis,
             const float* __restrict__ hx,
             float* __restrict__ agg) {
  __shared__ __align__(16) bf16_t tbuf[8][16 * DH];  // per-wave t tile  (32KB)
  __shared__ __align__(16) bf16_t hxb[8][16 * DH];   // gathered hx rows (32KB)
  const int wave = threadIdx.x >> 5, lane = threadIdx.x & 31;
  const int e0 = blockIdx.x * 128 + wave * 16;
  const int m = lane & 15, kh = (lane >> 4) & 1, n = lane & 15;
  bf16_t* tw = &tbuf[wave][0];
  bf16_t* hw = &hxb[wave][0];

  // ---- stage 0: coalesced gather of the wave's 16 source rows ----
#pragma unroll 4
  for (int e = 0; e < 16; ++e) {
    const int rr = eidx[e0 + e];                       // wave-uniform
    const float4 hv = *(const float4*)(hx + (size_t)rr * DH + lane * 4);
    v4bf p;
    p[0] = (bf16_t)hv.x; p[1] = (bf16_t)hv.y; p[2] = (bf16_t)hv.z; p[3] = (bf16_t)hv.w;
    *(v4bf*)(hw + e * DH + lane * 4) = p;
  }

  // ---- stage 1: A = edge_attr rows; per-lane real K is 8 contiguous floats
  const float4 a0 = *(const float4*)(ea + (size_t)(e0 + m) * DEA + kh * 8);
  const float4 a1v = *(const float4*)(ea + (size_t)(e0 + m) * DEA + kh * 8 + 4);
  v16bf a1 = {};
  a1[0] = (bf16_t)a0.x;  a1[1] = (bf16_t)a0.y;  a1[2] = (bf16_t)a0.z;  a1[3] = (bf16_t)a0.w;
  a1[4] = (bf16_t)a1v.x; a1[5] = (bf16_t)a1v.y; a1[6] = (bf16_t)a1v.z; a1[7] = (bf16_t)a1v.w;
  // elems 8..15 are K 16..31 -> zero pad (already zero-init)

#pragma unroll
  for (int t = 0; t < 8; ++t) {
    const v16bf b = W1p[t * 32 + lane];
    v8f c = {};
    c = WMMA_BF16(a1, b, c);
    const int feat = t * 16 + n;
    const float b1v = b1[feat];
#pragma unroll
    for (int r = 0; r < 8; ++r)
      tw[(r + (kh << 3)) * DH + feat] = (bf16_t)fmaxf(c[r] + b1v, 0.0f);
  }
  // same-wave LDS store->load: DScnt ordering, no workgroup barrier needed

  // ---- stage 2: ea = t @ W2, 4 K-chunks x 8 N-tiles ----
  v8f acc[8] = {};
#pragma unroll
  for (int kc = 0; kc < 4; ++kc) {
    const v16bf a = load_a_frag(tw + kc * 32, DH, lane);
#pragma unroll
    for (int t = 0; t < 8; ++t) {
      const v16bf b = W2p[(kc * 8 + t) * 32 + lane];   // 32B contiguous/lane
      acc[t] = WMMA_BF16(a, b, acc[t]);
    }
  }

  // ---- stage 3: msg = hx_row*w + ea + b2 ; scatter-add into agg[col] ----
  int colm[8];
  float wm[8];
#pragma unroll
  for (int r = 0; r < 8; ++r) {
    const int e = e0 + r + (kh << 3);
    const int rw = eidx[e];
    colm[r] = eidx[EE + e];
    wm[r] = dis[rw] * dis[colm[r]];
  }
#pragma unroll
  for (int t = 0; t < 8; ++t) {
    const int feat = t * 16 + n;
    const float b2v = b2[feat];
#pragma unroll
    for (int r = 0; r < 8; ++r) {
      const float hval = (float)hw[(r + (kh << 3)) * DH + feat];
      const float msg = hval * wm[r] + acc[t][r] + b2v;
      atomicAdd(&agg[(size_t)colm[r] * DH + feat], msg); // global_atomic_add_f32
    }
  }
}

// ---------------------------------------------------------------------------
// out = relu(LN(agg + bias) * ln_w + ln_b); one wave per node, 4 feats/lane.
// ---------------------------------------------------------------------------
__global__ void ln_relu_k(const float* __restrict__ agg,
                          const float* __restrict__ biasl,
                          const float* __restrict__ lnw,
                          const float* __restrict__ lnb,
                          float* __restrict__ dst) {
  const int wave = threadIdx.x >> 5, lane = threadIdx.x & 31;
  const int nid = blockIdx.x * 8 + wave;
  const int f = lane * 4;
  const float4 v4 = *(const float4*)(agg + (size_t)nid * DH + f);
  float v[4] = {v4.x + biasl[f], v4.y + biasl[f + 1],
                v4.z + biasl[f + 2], v4.w + biasl[f + 3]};
  float s = v[0] + v[1] + v[2] + v[3];
#pragma unroll
  for (int msk = 16; msk >= 1; msk >>= 1) s += __shfl_xor(s, msk, 32);
  const float mu = s * (1.0f / DH);
  float ss = 0.0f;
#pragma unroll
  for (int i = 0; i < 4; ++i) { v[i] -= mu; ss += v[i] * v[i]; }
#pragma unroll
  for (int msk = 16; msk >= 1; msk >>= 1) ss += __shfl_xor(ss, msk, 32);
  const float rinv = rsqrtf(ss * (1.0f / DH) + LN_EPS);
  float4 o;
  o.x = fmaxf(v[0] * rinv * lnw[f]     + lnb[f],     0.0f);
  o.y = fmaxf(v[1] * rinv * lnw[f + 1] + lnb[f + 1], 0.0f);
  o.z = fmaxf(v[2] * rinv * lnw[f + 2] + lnb[f + 2], 0.0f);
  o.w = fmaxf(v[3] * rinv * lnw[f + 3] + lnb[f + 3], 0.0f);
  *(float4*)(dst + (size_t)nid * DH + f) = o;
}

// ---------------------------------------------------------------------------
extern "C" void kernel_launch(void* const* d_in, const int* in_sizes, int n_in,
                              void* d_out, int out_size, void* d_ws, size_t ws_size,
                              hipStream_t stream) {
  const float* x        = (const float*)d_in[0];
  const float* edge_at  = (const float*)d_in[1];
  const float* lin_w    = (const float*)d_in[2];
  const float* edge_w1  = (const float*)d_in[3];
  const float* edge_b1  = (const float*)d_in[4];
  const float* edge_w2  = (const float*)d_in[5];
  const float* edge_b2  = (const float*)d_in[6];
  const float* bias     = (const float*)d_in[7];
  const float* ln_w     = (const float*)d_in[8];
  const float* ln_b     = (const float*)d_in[9];
  const int*   eidx     = (const int*)d_in[10];
  float* out = (float*)d_out;

  // workspace carve-up (256B aligned)
  char* ws = (char*)d_ws;
  size_t off = 0;
  auto carve = [&](size_t bytes) {
    void* p = ws + off;
    off += (bytes + 255) & ~(size_t)255;
    return p;
  };
  float*  dis  = (float*)carve((size_t)NN * 4);
  float*  hx   = (float*)carve((size_t)NN * DH * 4);
  float*  agg  = (float*)carve((size_t)NN * DH * 4);
  float*  hb0  = (float*)carve((size_t)NN * DH * 4);
  float*  hb1  = (float*)carve((size_t)NN * DH * 4);
  bf16_t* linp = (bf16_t*)carve((size_t)LAY * 4 * 8 * 32 * 16 * 2);
  bf16_t* w1p  = (bf16_t*)carve((size_t)LAY * 8 * 32 * 16 * 2);
  bf16_t* w2p  = (bf16_t*)carve((size_t)LAY * 4 * 8 * 32 * 16 * 2);

  // weights -> bf16 WMMA-fragment layout (once per call; deterministic)
  pack_b128_k<<<(LAY * 4 * 8 * 32 + 255) / 256, 256, 0, stream>>>(lin_w, linp);
  pack_b128_k<<<(LAY * 4 * 8 * 32 + 255) / 256, 256, 0, stream>>>(edge_w2, w2p);
  pack_b16_k<<<(LAY * 8 * 32 + 255) / 256, 256, 0, stream>>>(edge_w1, w1p);

  // gcn_norm: deg -> dis
  hipMemsetAsync(dis, 0, (size_t)NN * 4, stream);
  degree_k<<<(EE + 255) / 256, 256, 0, stream>>>(eidx + EE, dis);
  invsqrt_k<<<(NN + 255) / 256, 256, 0, stream>>>(dis);

  const float* hsrc[LAY] = {x, hb0, hb1};
  float*       hdst[LAY] = {hb0, hb1, out};

  for (int l = 0; l < LAY; ++l) {
    gemm_node_k<<<NN / 16, 256, 0, stream>>>(
        hsrc[l], (const v16bf*)(linp + (size_t)l * 4 * 8 * 32 * 16), hx);
    hipMemsetAsync(agg, 0, (size_t)NN * DH * 4, stream);
    edge_fused_k<<<EE / 128, 256, 0, stream>>>(
        edge_at,
        (const v16bf*)(w1p + (size_t)l * 8 * 32 * 16), edge_b1 + l * DH,
        (const v16bf*)(w2p + (size_t)l * 4 * 8 * 32 * 16), edge_b2 + l * DH,
        eidx, dis, hx, agg);
    ln_relu_k<<<NN / 8, 256, 0, stream>>>(
        agg, bias + l * DH, ln_w + l * DH, ln_b + l * DH, hdst[l]);
  }
}
